// MyRNN_20031727468783
// MI455X (gfx1250) — matrix operand order repaired and checked by reading, not compiled, not run on hardware
//
#include <hip/hip_runtime.h>
#include <hip/hip_bf16.h>

// ---------------------------------------------------------------------------
// 2-layer GRU (Keras reset_after=True), B=128, T=80, E=100, U=2048.
// Strategy: bf16 WMMA (v_wmma_f32_16x16x32_bf16) with f32 accumulate.
//  * Weights converted once per call into WMMA-B fragment order (bf16).
//  * Embedding rows gathered+packed into WMMA-A fragment order (K pad 100->128).
//  * One kernel launch per (t, layer); stream order = recurrence dependency.
//  * Each block owns 32 hidden columns and computes the z/r/h gate tiles for
//    those same columns, so the GRU pointwise math fuses in-kernel.
//  * h-state ping-pongs between two f32 buffers + two packed-bf16 A buffers.
// ---------------------------------------------------------------------------

#define BATCH  128
#define SEQ    80
#define EMB    100
#define UNITS  2048
#define GATES3 (3 * UNITS)       // 6144
#define KC_REC (UNITS / 32)      // 64 K-chunks for recurrent K=2048
#define KC_X   4                 // K=128 (EMB padded) for layer-0 input part
#define MT     (BATCH / 16)      // 8 M-tiles
#define NTILES (UNITS / 16)      // 128 N-tiles per gate
#define FRAG   512               // 32 lanes * 16 bf16 per fragment

typedef __attribute__((ext_vector_type(16))) __bf16 v16bf;
typedef __attribute__((ext_vector_type(8)))  float  v8f;
typedef unsigned short u16;

__device__ __forceinline__ u16 f32_bf16(float f) {
  union { float f; unsigned u; } v; v.f = f;
  unsigned r = v.u + 0x7FFFu + ((v.u >> 16) & 1u);   // round-to-nearest-even
  return (u16)(r >> 16);
}
__device__ __forceinline__ float sigmoidf_(float x) { return 1.f / (1.f + expf(-x)); }

// --- Pack f32 weight (K x 6144) into WMMA-B fragment order -----------------
// out[((g*NTILES + nt)*kchunks + kc)*FRAG + lane*16 + e]
//   lane<16: N=nt*16+lane,    K(local)=e<8 ? e : e+8      (0..7,16..23)
//   lane>=16:N=nt*16+lane-16, K(local)=e<8 ? 8+e : 16+e   (8..15,24..31)
__global__ void pack_weight_kernel(const float* __restrict__ W, u16* __restrict__ out,
                                   int Kact, int kchunks, long total) {
  long idx = (long)blockIdx.x * blockDim.x + threadIdx.x;
  if (idx >= total) return;
  int  e    = (int)(idx & 15);
  int  lane = (int)((idx >> 4) & 31);
  long rest = idx >> 9;
  int  kc   = (int)(rest % kchunks); rest /= kchunks;
  int  nt   = (int)(rest % NTILES);
  int  g    = (int)(rest / NTILES);
  int  n = g * UNITS + nt * 16 + (lane & 15);
  int  k = kc * 32 + (lane >> 4) * 8 + e + ((e < 8) ? 0 : 8);
  float v = (k < Kact) ? W[(long)k * GATES3 + n] : 0.f;
  out[idx] = f32_bf16(v);
}

// --- Gather embedding rows into WMMA-A fragment order (all timesteps) ------
// A 16-bit layout (ISA 7.12.2): lanes0-15 VGPR0..3 = K0..7, VGPR4..7 = K16..23
//                               lanes16-31 VGPR0..3 = K8..15, VGPR4..7 = K24..31
__global__ void pack_x_kernel(const int* __restrict__ tokens,
                              const float* __restrict__ emb,
                              u16* __restrict__ xp) {
  long idx = (long)blockIdx.x * blockDim.x + threadIdx.x;
  const long total = (long)SEQ * KC_X * MT * FRAG;
  if (idx >= total) return;
  int  e    = (int)(idx & 15);
  int  lane = (int)((idx >> 4) & 31);
  long rest = idx >> 9;
  int  mt = (int)(rest % MT); rest /= MT;
  int  kc = (int)(rest % KC_X);
  int  t  = (int)(rest / KC_X);
  int  hi = lane >> 4, mL = lane & 15;
  int  ko = e + ((e < 8) ? (hi ? 8 : 0) : (hi ? 16 : 8));
  int  k  = kc * 32 + ko;
  int  b  = mt * 16 + mL;
  float v = 0.f;
  if (k < EMB) {
    int tok = tokens[b * SEQ + t];
    v = emb[(long)tok * EMB + k];
  }
  xp[idx] = f32_bf16(v);
}

__global__ void zero_state_kernel(u16* h0p, u16* h1p, float* h0f, float* h1f, long n) {
  long i = (long)blockIdx.x * blockDim.x + threadIdx.x;
  if (i >= n) return;
  h0p[i] = 0; h1p[i] = 0; h0f[i] = 0.f; h1f[i] = 0.f;
}

// --- One GRU layer, one timestep -------------------------------------------
// gates = [A_in | A_rec] @ [W_in ; W_rec], z/r summed, h kept split (reset_after).
// Block: 256 thr = 8 waves (one 16-row M-tile each); grid: 64 blocks (32 cols each).
__global__ __launch_bounds__(256)
void gru_step_kernel(const u16* __restrict__ Ain,      // [kcIn][MT][FRAG]
                     const u16* __restrict__ Arec,     // [KC_REC][MT][FRAG]
                     const u16* __restrict__ Win,      // [3][NTILES][kcIn][FRAG]
                     const u16* __restrict__ Wrec,     // [3][NTILES][KC_REC][FRAG]
                     const float* __restrict__ bias,   // (2, 6144)
                     const float* __restrict__ hOld,   // (128, 2048) f32
                     float* __restrict__ hNew,         // (128, 2048) f32
                     u16* __restrict__ hNewPack,       // [KC_REC][MT][FRAG]
                     int kcIn) {
  const int lane = threadIdx.x & 31;
  const int wv   = threadIdx.x >> 5;       // M-tile index
  const int j0   = blockIdx.x * 32;        // hidden-column base
  const int nt0  = j0 >> 4;

  v8f zeroacc;
  #pragma unroll
  for (int i = 0; i < 8; ++i) zeroacc[i] = 0.f;
  v8f accz[2] = {zeroacc, zeroacc};
  v8f accr[2] = {zeroacc, zeroacc};
  v8f acci[2] = {zeroacc, zeroacc};        // h-gate, input contribution
  v8f acch[2] = {zeroacc, zeroacc};        // h-gate, recurrent contribution

  const long laneOff = (long)lane * 16;

  // ---- input-part GEMM: K = 32*kcIn ----
  {
    const u16* aP  = Ain + (long)wv * FRAG + laneOff;
    const u16* wz0 = Win + (((long)0 * NTILES + nt0 + 0) * kcIn) * FRAG + laneOff;
    const u16* wz1 = Win + (((long)0 * NTILES + nt0 + 1) * kcIn) * FRAG + laneOff;
    const u16* wr0 = Win + (((long)1 * NTILES + nt0 + 0) * kcIn) * FRAG + laneOff;
    const u16* wr1 = Win + (((long)1 * NTILES + nt0 + 1) * kcIn) * FRAG + laneOff;
    const u16* wh0 = Win + (((long)2 * NTILES + nt0 + 0) * kcIn) * FRAG + laneOff;
    const u16* wh1 = Win + (((long)2 * NTILES + nt0 + 1) * kcIn) * FRAG + laneOff;
    for (int kc = 0; kc < kcIn; ++kc) {
      const long ao = (long)kc * (MT * FRAG), wo = (long)kc * FRAG;
      v16bf a = *reinterpret_cast<const v16bf*>(aP + ao);
      v16bf b;
      b = *reinterpret_cast<const v16bf*>(wz0 + wo);
      accz[0] = __builtin_amdgcn_wmma_f32_16x16x32_bf16(false, a, false, b, (short)0, accz[0], false, false);
      b = *reinterpret_cast<const v16bf*>(wz1 + wo);
      accz[1] = __builtin_amdgcn_wmma_f32_16x16x32_bf16(false, a, false, b, (short)0, accz[1], false, false);
      b = *reinterpret_cast<const v16bf*>(wr0 + wo);
      accr[0] = __builtin_amdgcn_wmma_f32_16x16x32_bf16(false, a, false, b, (short)0, accr[0], false, false);
      b = *reinterpret_cast<const v16bf*>(wr1 + wo);
      accr[1] = __builtin_amdgcn_wmma_f32_16x16x32_bf16(false, a, false, b, (short)0, accr[1], false, false);
      b = *reinterpret_cast<const v16bf*>(wh0 + wo);
      acci[0] = __builtin_amdgcn_wmma_f32_16x16x32_bf16(false, a, false, b, (short)0, acci[0], false, false);
      b = *reinterpret_cast<const v16bf*>(wh1 + wo);
      acci[1] = __builtin_amdgcn_wmma_f32_16x16x32_bf16(false, a, false, b, (short)0, acci[1], false, false);
    }
  }

  // ---- recurrent-part GEMM: K = 2048 ----
  {
    const u16* aP  = Arec + (long)wv * FRAG + laneOff;
    const u16* wz0 = Wrec + (((long)0 * NTILES + nt0 + 0) * KC_REC) * FRAG + laneOff;
    const u16* wz1 = Wrec + (((long)0 * NTILES + nt0 + 1) * KC_REC) * FRAG + laneOff;
    const u16* wr0 = Wrec + (((long)1 * NTILES + nt0 + 0) * KC_REC) * FRAG + laneOff;
    const u16* wr1 = Wrec + (((long)1 * NTILES + nt0 + 1) * KC_REC) * FRAG + laneOff;
    const u16* wh0 = Wrec + (((long)2 * NTILES + nt0 + 0) * KC_REC) * FRAG + laneOff;
    const u16* wh1 = Wrec + (((long)2 * NTILES + nt0 + 1) * KC_REC) * FRAG + laneOff;
    for (int kc = 0; kc < KC_REC; ++kc) {
      const long ao = (long)kc * (MT * FRAG), wo = (long)kc * FRAG;
      v16bf a = *reinterpret_cast<const v16bf*>(aP + ao);
      v16bf b;
      b = *reinterpret_cast<const v16bf*>(wz0 + wo);
      accz[0] = __builtin_amdgcn_wmma_f32_16x16x32_bf16(false, a, false, b, (short)0, accz[0], false, false);
      b = *reinterpret_cast<const v16bf*>(wz1 + wo);
      accz[1] = __builtin_amdgcn_wmma_f32_16x16x32_bf16(false, a, false, b, (short)0, accz[1], false, false);
      b = *reinterpret_cast<const v16bf*>(wr0 + wo);
      accr[0] = __builtin_amdgcn_wmma_f32_16x16x32_bf16(false, a, false, b, (short)0, accr[0], false, false);
      b = *reinterpret_cast<const v16bf*>(wr1 + wo);
      accr[1] = __builtin_amdgcn_wmma_f32_16x16x32_bf16(false, a, false, b, (short)0, accr[1], false, false);
      b = *reinterpret_cast<const v16bf*>(wh0 + wo);
      acch[0] = __builtin_amdgcn_wmma_f32_16x16x32_bf16(false, a, false, b, (short)0, acch[0], false, false);
      b = *reinterpret_cast<const v16bf*>(wh1 + wo);
      acch[1] = __builtin_amdgcn_wmma_f32_16x16x32_bf16(false, a, false, b, (short)0, acch[1], false, false);
    }
  }

  // ---- fused GRU pointwise update ----
  // C-tile layout: VGPR v -> M = v + 8*(lane>=16), N = lane&15 (ISA 7.12.2)
  const int hi = lane >> 4;
  const int nl = lane & 15;
  const int m0 = wv * 16;
  #pragma unroll
  for (int s = 0; s < 2; ++s) {
    const int j   = j0 + s * 16 + nl;                       // hidden index
    const float bz  = bias[0 * UNITS + j] + bias[GATES3 + 0 * UNITS + j];
    const float br  = bias[1 * UNITS + j] + bias[GATES3 + 1 * UNITS + j];
    const float bhi = bias[2 * UNITS + j];
    const float bhh = bias[GATES3 + 2 * UNITS + j];
    const int kc = j >> 5;
    const int ko = j & 31;
    const int q  = ko >> 3;
    const int lp = 16 * (q & 1);
    const int ep = ko - ((q == 0) ? 0 : ((q == 3) ? 16 : 8));
    #pragma unroll
    for (int v = 0; v < 8; ++v) {
      const int   m  = m0 + v + 8 * hi;
      const float ho = hOld[(long)m * UNITS + j];
      const float z  = sigmoidf_(accz[s][v] + bz);
      const float r  = sigmoidf_(accr[s][v] + br);
      const float cd = tanhf(acci[s][v] + bhi + r * (acch[s][v] + bhh));
      const float hn = z * ho + (1.f - z) * cd;
      hNew[(long)m * UNITS + j] = hn;
      hNewPack[((long)(kc * MT + wv) * FRAG) + ((m & 15) + lp) * 16 + ep] = f32_bf16(hn);
    }
  }
}

// ---- final projection: sigmoid(h1 @ wout + bout) --------------------------
__global__ void out_kernel(const float* __restrict__ h1, const float* __restrict__ wout,
                           const float* __restrict__ bout, float* __restrict__ out) {
  int b = blockIdx.x * blockDim.x + threadIdx.x;
  if (b >= BATCH) return;
  float s = 0.f;
  for (int j = 0; j < UNITS; ++j) s += h1[(long)b * UNITS + j] * wout[j];
  s += bout[0];
  out[b] = 1.f / (1.f + expf(-s));
}

extern "C" void kernel_launch(void* const* d_in, const int* in_sizes, int n_in,
                              void* d_out, int out_size, void* d_ws, size_t ws_size,
                              hipStream_t stream) {
  const int*   tokens = (const int*)  d_in[0];
  const float* emb    = (const float*)d_in[1];
  const float* k0     = (const float*)d_in[2];
  const float* r0     = (const float*)d_in[3];
  const float* b0     = (const float*)d_in[4];
  const float* k1     = (const float*)d_in[5];
  const float* r1     = (const float*)d_in[6];
  const float* b1     = (const float*)d_in[7];
  const float* wout   = (const float*)d_in[8];
  const float* bout   = (const float*)d_in[9];
  float* out = (float*)d_out;

  const long SZ_K0P = 3L * NTILES * KC_X  * FRAG;   //   0.79M u16
  const long SZ_WRP = 3L * NTILES * KC_REC * FRAG;  //  12.58M u16 each
  const long SZ_XP  = (long)SEQ * KC_X * MT * FRAG; //   1.31M u16
  const long SZ_HP  = (long)KC_REC * MT * FRAG;     //   0.26M u16
  const long SZ_HF  = (long)BATCH * UNITS;          //   0.26M f32
  // total workspace ~86 MB

  u16* p = (u16*)d_ws;
  u16* k0p = p; p += SZ_K0P;
  u16* r0p = p; p += SZ_WRP;
  u16* k1p = p; p += SZ_WRP;
  u16* r1p = p; p += SZ_WRP;
  u16* xp  = p; p += SZ_XP;
  u16* h0p[2] = { p, p + SZ_HP }; p += 2 * SZ_HP;
  u16* h1p[2] = { p, p + SZ_HP }; p += 2 * SZ_HP;
  float* f = (float*)p;
  float* h0f[2] = { f, f + SZ_HF }; f += 2 * SZ_HF;
  float* h1f[2] = { f, f + SZ_HF };

  const dim3 blk(256);
  pack_weight_kernel<<<(unsigned)((SZ_K0P + 255) / 256), blk, 0, stream>>>(k0, k0p, EMB,   KC_X,   SZ_K0P);
  pack_weight_kernel<<<(unsigned)((SZ_WRP + 255) / 256), blk, 0, stream>>>(r0, r0p, UNITS, KC_REC, SZ_WRP);
  pack_weight_kernel<<<(unsigned)((SZ_WRP + 255) / 256), blk, 0, stream>>>(k1, k1p, UNITS, KC_REC, SZ_WRP);
  pack_weight_kernel<<<(unsigned)((SZ_WRP + 255) / 256), blk, 0, stream>>>(r1, r1p, UNITS, KC_REC, SZ_WRP);
  pack_x_kernel<<<(unsigned)((SZ_XP + 255) / 256), blk, 0, stream>>>(tokens, emb, xp);
  zero_state_kernel<<<(unsigned)((SZ_HP + 255) / 256), blk, 0, stream>>>(h0p[0], h1p[0], h0f[0], h1f[0], SZ_HP);

  for (int t = 0; t < SEQ; ++t) {
    const int cur = t & 1, nxt = 1 - cur;
    gru_step_kernel<<<UNITS / 32, blk, 0, stream>>>(
        xp + (long)t * KC_X * MT * FRAG, h0p[cur], k0p, r0p, b0,
        h0f[cur], h0f[nxt], h0p[nxt], KC_X);
    gru_step_kernel<<<UNITS / 32, blk, 0, stream>>>(
        h0p[nxt], h1p[cur], k1p, r1p, b1,
        h1f[cur], h1f[nxt], h1p[nxt], KC_REC);
  }
  out_kernel<<<1, 128, 0, stream>>>(h1f[SEQ & 1], wout, bout, out);

  (void)in_sizes; (void)n_in; (void)out_size; (void)ws_size;
}
